// KANDecoder_28329604284515
// MI455X (gfx1250) — compile-verified, matrix operand order checked by reading, not run on hardware
//
#include <hip/hip_runtime.h>
#include <hip/hip_bf16.h>

typedef __attribute__((ext_vector_type(16))) __bf16 v16bf;
typedef __attribute__((ext_vector_type(8)))  float  v8f;
typedef unsigned int u32x4 __attribute__((ext_vector_type(4)));
typedef int          i32x4 __attribute__((ext_vector_type(4)));
typedef int          i32x8 __attribute__((ext_vector_type(8)));

#define WMMA_BF16(a, b, c) \
  __builtin_amdgcn_wmma_f32_16x16x32_bf16(false, (a), false, (b), (short)0, (c), false, false)

// Load a 16x16x32 bf16 WMMA fragment for this lane from a 32-element K-row.
// CDNA5 layout: lane holds K-groups [(lane/16)*8, +8) and [16+(lane/16)*8, +8).
__device__ inline v16bf ld_frag16(const __bf16* base, int lane) {
  const int g = (lane >> 4) * 8;
  union { uint4 u[2]; v16bf v; } t;
  t.u[0] = *(const uint4*)(base + g);
  t.u[1] = *(const uint4*)(base + 16 + g);
  return t.v;
}

__device__ inline unsigned pack2bf(float a, float b) {
  union { __bf16 h[2]; unsigned u; } t;
  t.h[0] = (__bf16)a; t.h[1] = (__bf16)b;
  return t.u;
}

// SiLU with hardware v_rcp_f32 (avoids IEEE div expansion)
__device__ inline float silu_f(float x) {
  return x * __builtin_amdgcn_rcpf(1.0f + __expf(-x));
}

// ---------------------------------------------------------------------------
// Generic WMMA GEMM: C[M,N] = op(A)[M,K] @ W[N,K]^T (+bias) (+residual)
// M%64==0, N%64==0, K%32==0. 256 threads, 64x64 C tile, reg-prefetch k-loop.
// EPI: 0 = store f32, 1 = store f32 + residual, 2 = store bf16.
// ASILU: apply SiLU to A during staging.
// ---------------------------------------------------------------------------
struct Tile { float4 a0, a1, b0, b1; };

template <int EPI, bool ASILU>
__global__ __launch_bounds__(256) void gemm_wmma(
    const float* __restrict__ A, const float* __restrict__ W,
    const float* __restrict__ bias, const float* __restrict__ resid,
    float* __restrict__ C, __bf16* __restrict__ Cb, int M, int N, int K) {
  __shared__ __bf16 sA[64 * 48];
  __shared__ __bf16 sB[64 * 48];
  const int tid = threadIdx.x, lane = tid & 31, w = tid >> 5;
  const int mt = w & 3, nt = w >> 2;  // 4x2 wave grid: 16 rows x 32 cols each
  const int bm = blockIdx.y * 64, bn = blockIdx.x * 64;
  const int r = tid >> 2, c = (tid & 3) * 8;  // 8 consecutive elems per thread
  v8f acc0 = {0, 0, 0, 0, 0, 0, 0, 0};
  v8f acc1 = {0, 0, 0, 0, 0, 0, 0, 0};

  auto gload = [&](Tile& t, int k0) {
    const float* pa = A + (size_t)(bm + r) * K + k0 + c;
    const float* pw = W + (size_t)(bn + r) * K + k0 + c;
    t.a0 = *(const float4*)pa; t.a1 = *(const float4*)(pa + 4);
    t.b0 = *(const float4*)pw; t.b1 = *(const float4*)(pw + 4);
  };
  auto stage = [&](Tile& t) {
    float4 a0 = t.a0, a1 = t.a1;
    if (ASILU) {
      a0.x = silu_f(a0.x); a0.y = silu_f(a0.y); a0.z = silu_f(a0.z); a0.w = silu_f(a0.w);
      a1.x = silu_f(a1.x); a1.y = silu_f(a1.y); a1.z = silu_f(a1.z); a1.w = silu_f(a1.w);
    }
    uint4 ua = make_uint4(pack2bf(a0.x, a0.y), pack2bf(a0.z, a0.w),
                          pack2bf(a1.x, a1.y), pack2bf(a1.z, a1.w));
    uint4 ub = make_uint4(pack2bf(t.b0.x, t.b0.y), pack2bf(t.b0.z, t.b0.w),
                          pack2bf(t.b1.x, t.b1.y), pack2bf(t.b1.z, t.b1.w));
    *(uint4*)&sA[r * 48 + c] = ua;
    *(uint4*)&sB[r * 48 + c] = ub;
  };

  Tile cur, nxt;
  gload(cur, 0);
  for (int k0 = 0; k0 < K; k0 += 32) {
    if (k0) __syncthreads();  // previous compute done before LDS overwrite
    stage(cur);
    if (k0 + 32 < K) gload(nxt, k0 + 32);  // prefetch next tile into regs
    __syncthreads();
    v16bf af = ld_frag16(&sA[(mt * 16 + (lane & 15)) * 48], lane);
    v16bf b0 = ld_frag16(&sB[(nt * 32 + (lane & 15)) * 48], lane);
    v16bf b1 = ld_frag16(&sB[(nt * 32 + 16 + (lane & 15)) * 48], lane);
    acc0 = WMMA_BF16(af, b0, acc0);
    acc1 = WMMA_BF16(af, b1, acc1);
    cur = nxt;
  }
  const int half8 = (lane >> 4) * 8, nl = lane & 15;
#pragma unroll
  for (int rr = 0; rr < 8; ++rr) {
    int m = bm + mt * 16 + rr + half8;
    int n0 = bn + nt * 32 + nl, n1 = n0 + 16;
    float v0 = acc0[rr], v1 = acc1[rr];
    if (bias) { v0 += bias[n0]; v1 += bias[n1]; }
    size_t o0 = (size_t)m * N + n0, o1 = (size_t)m * N + n1;
    if (EPI == 2) {
      Cb[o0] = (__bf16)v0; Cb[o1] = (__bf16)v1;
    } else {
      if (EPI == 1) { v0 += resid[o0]; v1 += resid[o1]; }
      C[o0] = v0; C[o1] = v1;
    }
  }
}

// ---------------------------------------------------------------------------
// B-spline bases on the reference's UNIFORM grid (make_grid: t_j = -1+(j-3)h,
// h = 2/GRID_SIZE = 0.4). All Cox-de Boor denominators are p*h -> literal
// reciprocals; no divisions, no grid memory traffic.
// ---------------------------------------------------------------------------
__device__ inline void bsplines8(float x, float* __restrict__ out) {
  const float h = 0.4f;
  float t[12];
#pragma unroll
  for (int j = 0; j < 12; ++j) t[j] = -1.0f + (float)(j - 3) * h;
  float b[11];
#pragma unroll
  for (int j = 0; j < 11; ++j) b[j] = (x >= t[j] && x < t[j + 1]) ? 1.f : 0.f;
#pragma unroll
  for (int p = 1; p <= 3; ++p) {
    const float inv = 1.0f / ((float)p * h);  // compile-time constant
#pragma unroll
    for (int j = 0; j < 10; ++j) {
      if (j + p < 11) {
        b[j] = ((x - t[j]) * b[j] + (t[j + p + 1] - x) * b[j + 1]) * inv;
      }
    }
  }
#pragma unroll
  for (int j = 0; j < 8; ++j) out[j] = b[j];
}

// ---------------------------------------------------------------------------
// KAN spline GEMM with fused basis generation:
//   C[M,N] += bases(X)[M, IN*8] @ Ws[N, IN*8]^T
// A-tile: 4 features x 8 bases per K-chunk of 32, computed in-register.
// B-tile: staged by the Tensor Data Mover (tensor_load_to_lds), with D#
// padding producing the 48-elem LDS row stride directly.
// ---------------------------------------------------------------------------
__global__ __launch_bounds__(256) void kan_spline_wmma(
    const float* __restrict__ X, const __bf16* __restrict__ Ws,
    float* __restrict__ C, int M, int N, int IN) {
  __shared__ __bf16 sA[64 * 48];
  __shared__ __bf16 sB[64 * 48];
  const int tid = threadIdx.x, lane = tid & 31, w = tid >> 5;
  const int mt = w & 3, nt = w >> 2;
  const int bm = blockIdx.y * 64, bn = blockIdx.x * 64;
  const int K = IN * 8;
  v8f acc0 = {0, 0, 0, 0, 0, 0, 0, 0};
  v8f acc1 = {0, 0, 0, 0, 0, 0, 0, 0};
  const int rr = tid >> 2, ff = tid & 3;  // 64 rows x 4 features per chunk
  const unsigned ldsB = (unsigned)(uintptr_t)(void*)sB;

  for (int k0 = 0; k0 < K; k0 += 32) {
    if (k0) __syncthreads();
    // --- A tile: bases in registers -> one b128 LDS store ---
    {
      const int f = (k0 >> 3) + ff;
      float x = X[(size_t)(bm + rr) * IN + f];
      float bas[8];
      bsplines8(x, bas);
      uint4 ua = make_uint4(pack2bf(bas[0], bas[1]), pack2bf(bas[2], bas[3]),
                            pack2bf(bas[4], bas[5]), pack2bf(bas[6], bas[7]));
      *(uint4*)&sA[rr * 48 + ff * 8] = ua;
    }
    // --- B tile: TDM async copy 64x32 bf16 tile, padded to 48-elem rows ---
    if (tid < 32) {
      unsigned long long ga = (unsigned long long)(uintptr_t)(Ws + (size_t)bn * K + k0);
      union { unsigned d[4]; u32x4 v; } g0;
      g0.d[0] = 1u;                                    // count=1, user mode
      g0.d[1] = ldsB;                                  // lds_addr (bytes)
      g0.d[2] = (unsigned)ga;                          // global_addr[31:0]
      g0.d[3] = (unsigned)((ga >> 32) & 0x1FFFFFFu) | (2u << 30);  // addr[56:32], type=2
      union { int d[8]; i32x8 v; } g1;
      g1.d[0] = (1 << 16) | (1 << 20) | (3 << 22) | (7 << 25);
      // data_size=2B, pad_enable, pad_interval=16 dwords, pad_amount=8 dwords
      g1.d[1] = (K & 0xFFFF) << 16;                    // tensor_dim0 lo16
      g1.d[2] = ((K >> 16) & 0xFFFF) | ((N & 0xFFFF) << 16);  // dim0 hi / dim1 lo
      g1.d[3] = ((N >> 16) & 0xFFFF) | (32 << 16);     // dim1 hi / tile_dim0=32
      g1.d[4] = 64;                                    // tile_dim1=64, tile_dim2=0
      g1.d[5] = K;                                     // tensor_dim0_stride lo32
      g1.d[6] = 0;                                     // stride hi16 / dim1_stride lo16
      g1.d[7] = 0;
      i32x4 gz4 = {0, 0, 0, 0};
#if defined(__clang_major__) && __clang_major__ >= 23
      i32x8 gz8 = {0, 0, 0, 0, 0, 0, 0, 0};
      __builtin_amdgcn_tensor_load_to_lds(g0.v, g1.v, gz4, gz4, gz8, 0);
#else
      __builtin_amdgcn_tensor_load_to_lds(g0.v, g1.v, gz4, gz4, 0);
#endif
      __builtin_amdgcn_s_wait_tensorcnt((short)0);
    }
    __syncthreads();
    v16bf af = ld_frag16(&sA[(mt * 16 + (lane & 15)) * 48], lane);
    v16bf b0 = ld_frag16(&sB[(nt * 32 + (lane & 15)) * 48], lane);
    v16bf b1 = ld_frag16(&sB[(nt * 32 + 16 + (lane & 15)) * 48], lane);
    acc0 = WMMA_BF16(af, b0, acc0);
    acc1 = WMMA_BF16(af, b1, acc1);
  }
  const int half8 = (lane >> 4) * 8, nl = lane & 15;
#pragma unroll
  for (int r2 = 0; r2 < 8; ++r2) {
    int m = bm + mt * 16 + r2 + half8;
    int n0 = bn + nt * 32 + nl, n1 = n0 + 16;
    size_t o0 = (size_t)m * N + n0, o1 = (size_t)m * N + n1;
    C[o0] += acc0[r2];
    C[o1] += acc1[r2];
  }
}

// ---------------------------------------------------------------------------
// Multi-head attention, one block per (q-tile of 32, b*h).
// qkv bf16, layout (b, s, [q|k|v], h, d), row stride 1536.
// ---------------------------------------------------------------------------
__global__ __launch_bounds__(256) void attn_wmma(
    const __bf16* __restrict__ qkv, float* __restrict__ obuf, int S, int H) {
  extern __shared__ char smem[];
  __bf16* sQ  = (__bf16*)smem;            // 32 x 72  bf16  (4608 B)
  float*  sS  = (float*)(smem + 4608);    // 32 x 288 f32   (36864 B)
  __bf16* sP  = (__bf16*)(smem + 41472);  // 32 x 288 bf16  (18432 B)
  __bf16* sVt = (__bf16*)(smem + 59904);  // 64 x 288 bf16  (36864 B)
  const int tid = threadIdx.x, lane = tid & 31, w = tid >> 5;
  const int bh = blockIdx.y, b = bh / H, h = bh % H;
  const int q0 = blockIdx.x * 32;
  const int D3 = H * 64 * 3;  // 1536
  const size_t base = (size_t)b * S * D3 + h * 64;
  const uint4 z4 = make_uint4(0, 0, 0, 0);

  // Q tile: one b128 load + one b128 LDS store per thread
  {
    int r = tid >> 3, dg = tid & 7;
    int s = q0 + r;
    uint4 qv = (s < S) ? *(const uint4*)(qkv + base + (size_t)s * D3 + dg * 8) : z4;
    *(uint4*)&sQ[r * 72 + dg * 8] = qv;
  }
  // V^T: batch 9 vector loads into regs, then scatter bf16 to transposed LDS
  {
    uint4 vv[9];
#pragma unroll
    for (int i = 0; i < 9; ++i) {
      int idx = tid + i * 256;
      int dg = idx & 7, key = idx >> 3;
      vv[i] = (key < S) ? *(const uint4*)(qkv + base + (size_t)key * D3 + 1024 + dg * 8) : z4;
    }
#pragma unroll
    for (int i = 0; i < 9; ++i) {
      int idx = tid + i * 256;
      int dg = idx & 7, key = idx >> 3;
      union { uint4 u; __bf16 hh[8]; } t; t.u = vv[i];
#pragma unroll
      for (int j = 0; j < 8; ++j) sVt[(dg * 8 + j) * 288 + key] = t.hh[j];
    }
  }
  __syncthreads();

  // scores = Q K^T * 0.125 ; 17 key tiles of 16, Dh=64 -> 2 WMMA k-steps
  const int mt = w & 1;
  for (int ntile = (w >> 1); ntile < 17; ntile += 4) {
    v8f acc = {0, 0, 0, 0, 0, 0, 0, 0};
#pragma unroll
    for (int kk = 0; kk < 2; ++kk) {
      int k0 = kk * 32;
      v16bf af = ld_frag16(&sQ[(mt * 16 + (lane & 15)) * 72 + k0], lane);
      int key = ntile * 16 + (lane & 15);
      int ks = key < S ? key : S - 1;
      v16bf bf = ld_frag16(qkv + base + (size_t)ks * D3 + 512 + k0, lane);
      acc = WMMA_BF16(af, bf, acc);
    }
    int half8 = (lane >> 4) * 8, nl = lane & 15;
#pragma unroll
    for (int r = 0; r < 8; ++r) {
      int m = mt * 16 + r + half8;
      int n = ntile * 16 + nl;
      sS[m * 288 + n] = (n < S) ? acc[r] * 0.125f : -3.0e38f;
    }
  }
  __syncthreads();

  // softmax: 8 threads per row
  {
    int row = tid >> 3, sub = tid & 7;
    float mx = -3.0e38f;
    for (int c = sub; c < S; c += 8) mx = fmaxf(mx, sS[row * 288 + c]);
    for (int o = 1; o < 8; o <<= 1) mx = fmaxf(mx, __shfl_xor(mx, o, 32));
    float sum = 0.f;
    for (int c = sub; c < S; c += 8) {
      float e = __expf(sS[row * 288 + c] - mx);
      sS[row * 288 + c] = e;
      sum += e;
    }
    for (int o = 1; o < 8; o <<= 1) sum += __shfl_xor(sum, o, 32);
    float inv = __builtin_amdgcn_rcpf(sum);
    for (int c = sub; c < S; c += 8) sP[row * 288 + c] = (__bf16)(sS[row * 288 + c] * inv);
    for (int c = S + sub; c < 288; c += 8) sP[row * 288 + c] = (__bf16)0.f;
  }
  __syncthreads();

  // O = P @ V : waves tile (m 0..1) x (d-tile 0..3); K = 288 padded keys
  {
    int mt2 = w & 1, ntd = w >> 1;
    v8f acc = {0, 0, 0, 0, 0, 0, 0, 0};
    for (int k0 = 0; k0 < 288; k0 += 32) {
      v16bf af = ld_frag16(&sP[(mt2 * 16 + (lane & 15)) * 288 + k0], lane);
      v16bf bf = ld_frag16(&sVt[(ntd * 16 + (lane & 15)) * 288 + k0], lane);
      acc = WMMA_BF16(af, bf, acc);
    }
    int half8 = (lane >> 4) * 8, nl = lane & 15;
#pragma unroll
    for (int r = 0; r < 8; ++r) {
      int s = q0 + mt2 * 16 + r + half8;
      if (s < S) obuf[((size_t)b * S + s) * 512 + h * 64 + ntd * 16 + nl] = acc[r];
    }
  }
}

// ---------------------------------------------------------------------------
// LayerNorm over rows of length 512. One block (256 thr) per row.
// ---------------------------------------------------------------------------
__global__ __launch_bounds__(256) void layernorm_k(
    const float* __restrict__ X, const float* __restrict__ gw,
    const float* __restrict__ gb, float* __restrict__ Y) {
  const int row = blockIdx.x, tid = threadIdx.x, lane = tid & 31, w = tid >> 5;
  const float* xr = X + (size_t)row * 512;
  float* yr = Y + (size_t)row * 512;
  __shared__ float red[8];
  float v0 = xr[tid], v1 = xr[tid + 256];
  float s = v0 + v1;
  for (int o = 16; o > 0; o >>= 1) s += __shfl_xor(s, o, 32);
  if (lane == 0) red[w] = s;
  __syncthreads();
  float tot = 0.f;
#pragma unroll
  for (int i = 0; i < 8; ++i) tot += red[i];
  float mu = tot * (1.0f / 512.0f);
  __syncthreads();
  float d0 = v0 - mu, d1 = v1 - mu;
  float q = d0 * d0 + d1 * d1;
  for (int o = 16; o > 0; o >>= 1) q += __shfl_xor(q, o, 32);
  if (lane == 0) red[w] = q;
  __syncthreads();
  float vt = 0.f;
#pragma unroll
  for (int i = 0; i < 8; ++i) vt += red[i];
  float inv = rsqrtf(vt * (1.0f / 512.0f) + 1e-5f);
  yr[tid]       = d0 * inv * gw[tid] + gb[tid];
  yr[tid + 256] = d1 * inv * gw[tid + 256] + gb[tid + 256];
}

// ---------------------------------------------------------------------------
// Elementwise kernels
// ---------------------------------------------------------------------------
__global__ void unshuffle_k(const float* __restrict__ xe, const int* __restrict__ ids,
                            const float* __restrict__ mask, const float* __restrict__ pos,
                            float* __restrict__ out) {
  size_t idx = (size_t)blockIdx.x * 256 + threadIdx.x;
  const size_t n = (size_t)64 * 257 * 512;
  if (idx >= n) return;
  int b = (int)(idx / (257 * 512));
  int rem = (int)(idx % (257 * 512));
  int s = rem >> 9, d = rem & 511;
  float val;
  if (s == 0) {
    val = xe[((size_t)b * 65) * 512 + d];
  } else {
    int j = ids[b * 256 + (s - 1)];
    val = (j < 64) ? xe[((size_t)b * 65 + 1 + j) * 512 + d] : mask[d];
  }
  out[idx] = val + pos[(size_t)s * 512 + d];
}

__global__ void scale_w_k(const float* __restrict__ sw, const float* __restrict__ sc,
                          __bf16* __restrict__ out, size_t n) {
  size_t i = (size_t)blockIdx.x * 256 + threadIdx.x;
  if (i >= n) return;
  out[i] = (__bf16)(sw[i] * sc[i >> 3]);
}

__global__ void slice_out_k(const float* __restrict__ G, float* __restrict__ out) {
  size_t idx = (size_t)blockIdx.x * 256 + threadIdx.x;
  const size_t n = (size_t)64 * 256 * 256;
  if (idx >= n) return;
  int b = (int)(idx >> 16);
  int rem = (int)(idx & 65535);
  int s = rem >> 8, d = rem & 255;
  out[idx] = G[((size_t)b * 257 + s + 1) * 256 + d];
}

// ---------------------------------------------------------------------------
// Host launch
// ---------------------------------------------------------------------------
extern "C" void kernel_launch(void* const* d_in, const int* in_sizes, int n_in,
                              void* d_out, int out_size, void* d_ws, size_t ws_size,
                              hipStream_t stream) {
  (void)in_sizes; (void)n_in; (void)out_size; (void)ws_size;
  const float* x_in  = (const float*)d_in[0];
  const int*   ids   = (const int*)d_in[1];
  const float* W_emb = (const float*)d_in[2];
  const float* b_emb = (const float*)d_in[3];
  const float* maskt = (const float*)d_in[4];
  const float* pos   = (const float*)d_in[5];
  const float* ln1w  = (const float*)d_in[6];
  const float* ln1b  = (const float*)d_in[7];
  const float* wqkv  = (const float*)d_in[8];
  const float* bqkv  = (const float*)d_in[9];
  const float* wout  = (const float*)d_in[10];
  const float* bout  = (const float*)d_in[11];
  const float* ln2w  = (const float*)d_in[12];
  const float* ln2b  = (const float*)d_in[13];
  const float* k1bw  = (const float*)d_in[14];
  const float* k1sw  = (const float*)d_in[15];
  const float* k1sc  = (const float*)d_in[16];
  const float* k2bw  = (const float*)d_in[18];
  const float* k2sw  = (const float*)d_in[19];
  const float* k2sc  = (const float*)d_in[20];
  float* out = (float*)d_out;

  const size_t ME = (size_t)4160 * 512;    // embed out elems
  const size_t MD = (size_t)16448 * 512;   // (B*S, 512) elems
  const size_t MQ = (size_t)16448 * 1536;  // qkv elems
  const size_t MG = (size_t)16448 * 256;   // kan2 out elems

  auto aup = [](size_t v) { return (v + 255) & ~(size_t)255; };
  char* p = (char*)d_ws;
  float*  bufA = (float*)p;  p += aup(ME * 4);   // embed out
  float*  bufX = (float*)p;  p += aup(MD * 4);   // residual stream
  float*  bufC = (float*)p;  p += aup(MD * 4);   // ln2 out
  __bf16* qkvb = (__bf16*)p; p += aup(MQ * 2);   // qkv bf16 (later: kan1 out f32)
  float*  bufE = (float*)p;  p += aup(MD * 4);   // ln1 out / attn o
  __bf16* ws1  = (__bf16*)p; p += aup((size_t)512 * 4096 * 2);
  __bf16* ws2  = (__bf16*)p; p += aup((size_t)256 * 4096 * 2);
  float*  bufG = (float*)p;  p += aup(MG * 4);
  float* K1 = (float*)qkvb;  // kan1 out f32 overlays dead qkv bf16

  dim3 blk(256);

  // 1) embed: (4160,384) @ (512,384)^T + b
  gemm_wmma<0, false><<<dim3(8, 65), blk, 0, stream>>>(
      x_in, W_emb, b_emb, nullptr, bufA, nullptr, 4160, 512, 384);
  // 2) unshuffle + mask token + pos embed -> residual stream
  {
    size_t n = (size_t)64 * 257 * 512;
    unshuffle_k<<<(unsigned)((n + 255) / 256), blk, 0, stream>>>(bufA, ids, maskt, pos, bufX);
  }
  // 3) LN1
  layernorm_k<<<16448, blk, 0, stream>>>(bufX, ln1w, ln1b, bufE);
  // 4) QKV GEMM -> bf16 directly (no separate convert pass)
  gemm_wmma<2, false><<<dim3(24, 257), blk, 0, stream>>>(
      bufE, wqkv, bqkv, nullptr, nullptr, qkvb, 16448, 1536, 512);
  // 5) attention (9 q-tiles x 512 (b,h) pairs), 96768 B dynamic LDS
  attn_wmma<<<dim3(9, 512), blk, 96768, stream>>>(qkvb, bufE, 257, 8);
  // 6) out_proj + residual
  gemm_wmma<1, false><<<dim3(8, 257), blk, 0, stream>>>(
      bufE, wout, bout, bufX, bufX, nullptr, 16448, 512, 512);
  // 7) LN2
  layernorm_k<<<16448, blk, 0, stream>>>(bufX, ln2w, ln2b, bufC);
  // 8) prepack scaled spline weights (bf16)
  scale_w_k<<<(unsigned)(((size_t)512 * 4096 + 255) / 256), blk, 0, stream>>>(
      k1sw, k1sc, ws1, (size_t)512 * 4096);
  scale_w_k<<<(unsigned)(((size_t)256 * 4096 + 255) / 256), blk, 0, stream>>>(
      k2sw, k2sc, ws2, (size_t)256 * 4096);
  // 9) KAN1 base GEMM, SiLU fused into A staging
  gemm_wmma<0, true><<<dim3(8, 257), blk, 0, stream>>>(
      bufC, k1bw, nullptr, nullptr, K1, nullptr, 16448, 512, 512);
  // 10) KAN1 spline GEMM (fused bases + TDM weight staging, accumulate)
  kan_spline_wmma<<<dim3(8, 257), blk, 0, stream>>>(bufC, ws1, K1, 16448, 512, 512);
  // 11) KAN2 base GEMM, SiLU fused into A staging
  gemm_wmma<0, true><<<dim3(4, 257), blk, 0, stream>>>(
      K1, k2bw, nullptr, nullptr, bufG, nullptr, 16448, 256, 512);
  // 12) KAN2 spline GEMM (accumulate)
  kan_spline_wmma<<<dim3(4, 257), blk, 0, stream>>>(K1, ws2, bufG, 16448, 256, 512);
  // 13) drop cls row, write (64,256,256)
  {
    size_t n = (size_t)64 * 256 * 256;
    slice_out_k<<<(unsigned)((n + 255) / 256), blk, 0, stream>>>(bufG, out);
  }
}
